// ClinicalStateFormationOperator_54296976556726
// MI455X (gfx1250) — matrix operand order repaired and checked by reading, not compiled, not run on hardware
//
#include <hip/hip_runtime.h>

typedef __attribute__((ext_vector_type(16))) _Float16 v16h;
typedef __attribute__((ext_vector_type(8)))  _Float16 v8h;
typedef __attribute__((ext_vector_type(8)))  float    v8f;
typedef __attribute__((ext_vector_type(4)))  int      v4i;
typedef __attribute__((address_space(1))) v4i g_v4i;   // global-side payload ptr for async DMA
typedef __attribute__((address_space(3))) v4i l_v4i;   // LDS-side payload ptr for async DMA

#define Bc 4
#define Tt 48
#define Vv 32
#define Dd 512
#define Hh 8
#define HD 64
#define Nn (Tt*Vv)   /* 1536 */
#define QK 96        /* 64 content dims + 16 obs dims + 16 zero pad */
#define KLD 112      /* LDS row stride (halves) for K panel */
#define VLD 40       /* LDS row stride (halves) for V/A/B panels */

#if defined(__has_builtin)
#if __has_builtin(__builtin_amdgcn_global_load_async_to_lds_b128)
#define USE_ASYNC 1
#else
#define USE_ASYNC 0
#endif
#else
#define USE_ASYNC 0
#endif

union FragH { v16h v; v8h h[2]; };

__device__ __forceinline__ v8f wmma16(v16h a, v16h b, v8f c) {
  return __builtin_amdgcn_wmma_f32_16x16x32_f16(false, a, false, b, (short)0, c, false, false);
}

// copy 16B global -> LDS (async DMA on gfx1250 when available, sync fallback otherwise)
__device__ __forceinline__ void cp16(const _Float16* g, _Float16* l) {
#if USE_ASYNC
  __builtin_amdgcn_global_load_async_to_lds_b128((g_v4i*)g, (l_v4i*)l, 0, 0);
#else
  *(v8h*)l = *(const v8h*)g;
#endif
}
__device__ __forceinline__ void async_wait0() {
#if USE_ASYNC
#if __has_builtin(__builtin_amdgcn_s_wait_asynccnt)
  __builtin_amdgcn_s_wait_asynccnt(0);
#else
  asm volatile("s_wait_asynccnt 0x0" ::: "memory");
#endif
#endif
}

// A fragment from global, row-major: lane row M=lane%16; K chunks half*8 and 16+half*8
__device__ __forceinline__ v16h load_a_frag(const _Float16* __restrict__ base, int ld, int row0, int k0, int lane) {
  int M = lane & 15, half = lane >> 4;
  const _Float16* p = base + (size_t)(row0 + M) * ld + k0 + half * 8;
  FragH f;
  f.h[0] = *(const v8h*)(p);
  f.h[1] = *(const v8h*)(p + 16);
  return f.v;
}

// ---- f32 -> f16 conversion of h ----
__global__ void k_cvt_h(const float* __restrict__ src, _Float16* __restrict__ dst, int n) {
  int i = (blockIdx.x * blockDim.x + threadIdx.x) * 4;
  if (i + 3 < n) {
    float4 f = *(const float4*)(src + i);
    dst[i+0] = (_Float16)f.x; dst[i+1] = (_Float16)f.y;
    dst[i+2] = (_Float16)f.z; dst[i+3] = (_Float16)f.w;
  }
}

// ---- transpose + convert four DxD weights: wT[m][out][in] = W[m][in][out] ----
__global__ void k_cvt_wT(const float* __restrict__ W0, const float* __restrict__ W1,
                         const float* __restrict__ W2, const float* __restrict__ W3,
                         _Float16* __restrict__ dst) {
  int idx = blockIdx.x * blockDim.x + threadIdx.x;
  int m = idx >> 18;
  int r = idx & (Dd*Dd - 1);
  int o = r & (Dd - 1);
  int i = r >> 9;
  const float* W = (m == 0) ? W0 : (m == 1) ? W1 : (m == 2) ? W2 : W3;
  dst[(size_t)m*Dd*Dd + (size_t)o*Dd + i] = (_Float16)W[(size_t)i*Dd + o];
}

// ---- tiled QKV GEMM: 128x128 block tile, 8 waves x (32 rows x 64 cols) ----
__global__ void k_qkv_t(const _Float16* __restrict__ hf,
                        const _Float16* __restrict__ wT,
                        const float* __restrict__ bq, const float* __restrict__ bk,
                        const float* __restrict__ bv,
                        _Float16* __restrict__ qcat, _Float16* __restrict__ kcat,
                        _Float16* __restrict__ vT) {
  __shared__ __align__(16) _Float16 abuf[128 * VLD];
  __shared__ __align__(16) _Float16 bbuf[128 * VLD];
  int tid = threadIdx.x;
  int lane = tid & 31, w = tid >> 5;
  int m = blockIdx.x / 96;
  int rem = blockIdx.x % 96;
  int r0 = (rem >> 2) * 128;
  int c0 = (rem & 3) * 128;
  int wr = w >> 1, wc = w & 1;
  int M = lane & 15, half = lane >> 4, nl = lane & 15;
  const _Float16* wmat = wT + (size_t)m * Dd * Dd;

  v8f acc[2][4];
  v8f zero = {};
  for (int s = 0; s < 2; ++s) for (int t = 0; t < 4; ++t) acc[s][t] = zero;

  int srow = tid >> 2, sc = tid & 3;            // staging: thread -> (row, 16B chunk)
  for (int k0 = 0; k0 < Dd; k0 += 32) {
    __syncthreads();                            // WAR: previous iteration's reads done
    cp16(hf   + (size_t)(r0 + srow)      *Dd + k0 + sc*8, &abuf[ srow      *VLD + sc*8]);
    cp16(hf   + (size_t)(r0 + srow + 64) *Dd + k0 + sc*8, &abuf[(srow + 64)*VLD + sc*8]);
    cp16(wmat + (size_t)(c0 + srow)      *Dd + k0 + sc*8, &bbuf[ srow      *VLD + sc*8]);
    cp16(wmat + (size_t)(c0 + srow + 64) *Dd + k0 + sc*8, &bbuf[(srow + 64)*VLD + sc*8]);
    async_wait0();
    __syncthreads();
    FragH a0, a1;
    a0.h[0] = *(const v8h*)&abuf[(wr*32 + M)*VLD + half*8];
    a0.h[1] = *(const v8h*)&abuf[(wr*32 + M)*VLD + 16 + half*8];
    a1.h[0] = *(const v8h*)&abuf[(wr*32 + 16 + M)*VLD + half*8];
    a1.h[1] = *(const v8h*)&abuf[(wr*32 + 16 + M)*VLD + 16 + half*8];
    for (int t = 0; t < 4; ++t) {
      FragH fb;
      int col = wc*64 + t*16 + nl;
      fb.h[0] = *(const v8h*)&bbuf[col*VLD + half*8];
      fb.h[1] = *(const v8h*)&bbuf[col*VLD + 16 + half*8];
      acc[0][t] = wmma16(a0.v, fb.v, acc[0][t]);
      acc[1][t] = wmma16(a1.v, fb.v, acc[1][t]);
    }
  }

  const float* bias = (m == 0) ? bq : (m == 1) ? bk : bv;
  for (int t = 0; t < 4; ++t) {
    int d = c0 + wc*64 + t*16 + nl;
    int hIdx = d >> 6, hd = d & 63;
    float bb = bias[d];
    for (int s = 0; s < 2; ++s) {
      for (int r = 0; r < 8; ++r) {
        int row = r0 + wr*32 + s*16 + r + 8*half;
        int b_ = row / Nn, n = row % Nn;
        float val = acc[s][t][r] + bb;
        if (m == 0)      qcat[((size_t)(b_*Hh + hIdx)*Nn + n)*QK + hd] = (_Float16)(val * 0.125f);
        else if (m == 1) kcat[((size_t)(b_*Hh + hIdx)*Nn + n)*QK + hd] = (_Float16)val;
        else             vT[((size_t)(b_*Hh + hIdx)*HD + hd)*Nn + n]   = (_Float16)val;
      }
    }
  }
}

// ---- obs-state projections into qcat/kcat cols 64..95 ----
__global__ void k_obs(const float* __restrict__ fo,
                      const float* __restrict__ Woq, const float* __restrict__ boq,
                      const float* __restrict__ Wok, const float* __restrict__ bok,
                      _Float16* __restrict__ qcat, _Float16* __restrict__ kcat) {
  int idx = blockIdx.x * blockDim.x + threadIdx.x;
  if (idx >= Bc*Nn*Hh*32) return;
  int cpos = idx & 31;
  int hIdx = (idx >> 5) & 7;
  int bn = idx >> 8;
  size_t rowbase = ((size_t)((bn / Nn)*Hh + hIdx)*Nn + (bn % Nn))*QK;
  if (cpos < 16) {
    int j = hIdx*16 + cpos;
    float f0 = fo[bn*2 + 0], f1 = fo[bn*2 + 1];
    float oq = f0*Woq[j] + f1*Woq[128 + j] + boq[j];
    float ok = f0*Wok[j] + f1*Wok[128 + j] + bok[j];
    qcat[rowbase + 64 + cpos] = (_Float16)(oq * 0.25f);
    kcat[rowbase + 64 + cpos] = (_Float16)ok;
  } else {
    qcat[rowbase + 64 + cpos] = (_Float16)0.f;
    kcat[rowbase + 64 + cpos] = (_Float16)0.f;
  }
}

// ---- fused attention: block = 8 query-tile waves sharing one (b,h); K/V staged in LDS ----
__global__ void k_attn(const _Float16* __restrict__ qcat, const _Float16* __restrict__ kcat,
                       const _Float16* __restrict__ vT,
                       const float* __restrict__ var_bias,
                       const float* __restrict__ time_bias,
                       _Float16* __restrict__ att_out) {
  __shared__ __align__(16) _Float16 kbuf[32 * KLD];
  __shared__ __align__(16) _Float16 vbuf[64 * VLD];
  __shared__ __align__(16) _Float16 pst[8][16*32];
  int tid = threadIdx.x;
  int lane = tid & 31, w = tid >> 5;
  int bh = blockIdx.x / 12;
  int qt = (blockIdx.x % 12) * 8 + w;
  int b_ = bh / Hh, hIdx = bh % Hh;
  int half = lane >> 4, nl = lane & 15;
  const _Float16* qb = qcat + (size_t)bh * Nn * QK;
  const _Float16* kb = kcat + (size_t)bh * Nn * QK;
  const _Float16* vb = vT   + (size_t)bh * HD * Nn;
  const float* vbias = var_bias + hIdx*Vv*Vv;
  const float* tbias = time_bias + hIdx*(2*Tt - 1);

  v16h aq0 = load_a_frag(qb, QK, qt*16, 0,  lane);
  v16h aq1 = load_a_frag(qb, QK, qt*16, 32, lane);
  v16h aq2 = load_a_frag(qb, QK, qt*16, 64, lane);

  float mrow[8], lrow[8];
  v8f acc[4];
  v8f zero = {};
  for (int r = 0; r < 8; ++r) { mrow[r] = -3.0e38f; lrow[r] = 0.f; }
  for (int t = 0; t < 4; ++t) acc[t] = zero;

  int tq[8], vq[8];
  for (int r = 0; r < 8; ++r) {
    int n = qt*16 + r + 8*half;
    tq[r] = n / Vv; vq[r] = n % Vv;
  }

  int krow = tid / 12, kc = tid % 12;                     // K-panel chunks 0..255
  int krow2 = (tid + 256) / 12, kc2 = (tid + 256) % 12;   // chunks 256..383 (tid<128)
  int vrow = tid >> 2, vc = tid & 3;                      // V-panel chunks

  for (int kbk = 0; kbk < Nn/32; ++kbk) {
    __syncthreads();                         // WAR for kbuf/vbuf/pst
    cp16(kb + (size_t)(kbk*32 + krow)*QK + kc*8, &kbuf[krow*KLD + kc*8]);
    if (tid < 128)
      cp16(kb + (size_t)(kbk*32 + krow2)*QK + kc2*8, &kbuf[krow2*KLD + kc2*8]);
    cp16(vb + (size_t)vrow*Nn + kbk*32 + vc*8, &vbuf[vrow*VLD + vc*8]);
    if (kbk + 1 < Nn/32) {                   // prefetch next block toward WGP
      __builtin_prefetch(kb + (size_t)((kbk+1)*32 + (tid & 31))*QK, 0, 1);
      __builtin_prefetch(vb + (size_t)(tid & 63)*Nn + (kbk+1)*32, 0, 1);
    }
    async_wait0();
    __syncthreads();

    v8f s[2];
    for (int sub = 0; sub < 2; ++sub) {
      v8f sc2 = {};
      for (int c3 = 0; c3 < 3; ++c3) {
        FragH fb;
        int koff = (sub*16 + nl)*KLD + c3*32 + half*8;
        fb.h[0] = *(const v8h*)&kbuf[koff];
        fb.h[1] = *(const v8h*)&kbuf[koff + 16];
        v16h aq = (c3 == 0) ? aq0 : (c3 == 1) ? aq1 : aq2;
        sc2 = wmma16(aq, fb.v, sc2);
      }
      int nk = (kbk*2 + sub)*16 + nl;
      int tk = nk / Vv, vk = nk % Vv;
      for (int r = 0; r < 8; ++r)
        sc2[r] += vbias[vq[r]*Vv + vk] + tbias[tq[r] - tk + (Tt - 1)];
      s[sub] = sc2;
    }

    float alpha[8];
    for (int r = 0; r < 8; ++r) {
      float tm = fmaxf(s[0][r], s[1][r]);
      for (int off = 8; off > 0; off >>= 1) tm = fmaxf(tm, __shfl_xor(tm, off, 16));
      float mn = fmaxf(mrow[r], tm);
      alpha[r] = __expf(mrow[r] - mn);
      mrow[r] = mn;
      float p0 = __expf(s[0][r] - mn);
      float p1 = __expf(s[1][r] - mn);
      float ls = p0 + p1;
      for (int off = 8; off > 0; off >>= 1) ls += __shfl_xor(ls, off, 16);
      lrow[r] = lrow[r]*alpha[r] + ls;
      int prow = r + 8*half;
      pst[w][prow*32 + nl]      = (_Float16)p0;
      pst[w][prow*32 + 16 + nl] = (_Float16)p1;
    }
    for (int t = 0; t < 4; ++t)
      for (int r = 0; r < 8; ++r) acc[t][r] *= alpha[r];
    __syncthreads();                         // pst visible wave-wide

    FragH ap;
    ap.h[0] = *(const v8h*)&pst[w][nl*32 + half*8];
    ap.h[1] = *(const v8h*)&pst[w][nl*32 + 16 + half*8];
    for (int t = 0; t < 4; ++t) {
      FragH fv;
      int voff = (t*16 + nl)*VLD + half*8;
      fv.h[0] = *(const v8h*)&vbuf[voff];
      fv.h[1] = *(const v8h*)&vbuf[voff + 16];
      acc[t] = wmma16(ap.v, fv.v, acc[t]);
    }
  }

  float inv[8];
  for (int r = 0; r < 8; ++r) inv[r] = 1.f / lrow[r];
  for (int t = 0; t < 4; ++t) {
    int d = hIdx*HD + t*16 + nl;
    for (int r = 0; r < 8; ++r) {
      int n = qt*16 + r + 8*half;
      att_out[(size_t)(b_*Nn + n)*Dd + d] = (_Float16)(acc[t][r] * inv[r]);
    }
  }
}

// ---- tiled output projection: att_out @ Wo + bo -> fp32 ----
__global__ void k_out_t(const _Float16* __restrict__ ao, const _Float16* __restrict__ woT,
                        const float* __restrict__ bo, float* __restrict__ out) {
  __shared__ __align__(16) _Float16 abuf[128 * VLD];
  __shared__ __align__(16) _Float16 bbuf[128 * VLD];
  int tid = threadIdx.x;
  int lane = tid & 31, w = tid >> 5;
  int r0 = (blockIdx.x >> 2) * 128;
  int c0 = (blockIdx.x & 3) * 128;
  int wr = w >> 1, wc = w & 1;
  int M = lane & 15, half = lane >> 4, nl = lane & 15;

  v8f acc[2][4];
  v8f zero = {};
  for (int s = 0; s < 2; ++s) for (int t = 0; t < 4; ++t) acc[s][t] = zero;

  int srow = tid >> 2, sc = tid & 3;
  for (int k0 = 0; k0 < Dd; k0 += 32) {
    __syncthreads();
    cp16(ao  + (size_t)(r0 + srow)      *Dd + k0 + sc*8, &abuf[ srow      *VLD + sc*8]);
    cp16(ao  + (size_t)(r0 + srow + 64) *Dd + k0 + sc*8, &abuf[(srow + 64)*VLD + sc*8]);
    cp16(woT + (size_t)(c0 + srow)      *Dd + k0 + sc*8, &bbuf[ srow      *VLD + sc*8]);
    cp16(woT + (size_t)(c0 + srow + 64) *Dd + k0 + sc*8, &bbuf[(srow + 64)*VLD + sc*8]);
    async_wait0();
    __syncthreads();
    FragH a0, a1;
    a0.h[0] = *(const v8h*)&abuf[(wr*32 + M)*VLD + half*8];
    a0.h[1] = *(const v8h*)&abuf[(wr*32 + M)*VLD + 16 + half*8];
    a1.h[0] = *(const v8h*)&abuf[(wr*32 + 16 + M)*VLD + half*8];
    a1.h[1] = *(const v8h*)&abuf[(wr*32 + 16 + M)*VLD + 16 + half*8];
    for (int t = 0; t < 4; ++t) {
      FragH fb;
      int col = wc*64 + t*16 + nl;
      fb.h[0] = *(const v8h*)&bbuf[col*VLD + half*8];
      fb.h[1] = *(const v8h*)&bbuf[col*VLD + 16 + half*8];
      acc[0][t] = wmma16(a0.v, fb.v, acc[0][t]);
      acc[1][t] = wmma16(a1.v, fb.v, acc[1][t]);
    }
  }

  for (int t = 0; t < 4; ++t) {
    int d = c0 + wc*64 + t*16 + nl;
    float bb = bo[d];
    for (int s = 0; s < 2; ++s) {
      for (int r = 0; r < 8; ++r) {
        int row = r0 + wr*32 + s*16 + r + 8*half;
        out[(size_t)row*Dd + d] = acc[s][t][r] + bb;
      }
    }
  }
}

extern "C" void kernel_launch(void* const* d_in, const int* in_sizes, int n_in,
                              void* d_out, int out_size, void* d_ws, size_t ws_size,
                              hipStream_t stream) {
  (void)in_sizes; (void)n_in; (void)out_size; (void)ws_size;
  const float* h   = (const float*)d_in[0];
  const float* fo  = (const float*)d_in[1];
  const float* Wq  = (const float*)d_in[2];
  const float* bq  = (const float*)d_in[3];
  const float* Wk  = (const float*)d_in[4];
  const float* bk  = (const float*)d_in[5];
  const float* Wv  = (const float*)d_in[6];
  const float* bv  = (const float*)d_in[7];
  const float* Wo  = (const float*)d_in[8];
  const float* bo  = (const float*)d_in[9];
  const float* Woq = (const float*)d_in[10];
  const float* boq = (const float*)d_in[11];
  const float* Wok = (const float*)d_in[12];
  const float* bok = (const float*)d_in[13];
  const float* vbias = (const float*)d_in[14];
  const float* tbias = (const float*)d_in[15];
  float* out = (float*)d_out;

  _Float16* ws = (_Float16*)d_ws;
  const size_t nBND = (size_t)Bc*Nn*Dd;
  _Float16* hf16   = ws;                           // reused as att_out after k_qkv_t
  _Float16* wT     = hf16 + nBND;
  _Float16* qcat   = wT + (size_t)4*Dd*Dd;
  _Float16* kcat   = qcat + (size_t)Bc*Hh*Nn*QK;
  _Float16* vT     = kcat + (size_t)Bc*Hh*Nn*QK;
  _Float16* attout = hf16;
  _Float16* woT    = wT + (size_t)3*Dd*Dd;

  k_cvt_h<<<(int)(nBND/4/256), 256, 0, stream>>>(h, hf16, (int)nBND);
  k_cvt_wT<<<4*Dd*Dd/256, 256, 0, stream>>>(Wq, Wk, Wv, Wo, wT);
  k_qkv_t<<<3*24*4, 256, 0, stream>>>(hf16, wT, bq, bk, bv, qcat, kcat, vT);
  k_obs<<<(Bc*Nn*Hh*32)/256, 256, 0, stream>>>(fo, Woq, boq, Wok, bok, qcat, kcat);
  k_attn<<<Bc*Hh*12, 256, 0, stream>>>(qcat, kcat, vT, vbias, tbias, attout);
  k_out_t<<<24*4, 256, 0, stream>>>(attout, woT, bo, out);
}
